// LinearAttention_5549097747050
// MI455X (gfx1250) — matrix operand order repaired and checked by reading, not compiled, not run on hardware
//
#include <hip/hip_runtime.h>
#include <hip/hip_bf16.h>

typedef _Float16 h16;
typedef __attribute__((ext_vector_type(16))) _Float16 v16h;
typedef __attribute__((ext_vector_type(8)))  _Float16 v8h;
typedef __attribute__((ext_vector_type(4)))  _Float16 v4h;
typedef __attribute__((ext_vector_type(2)))  _Float16 v2h;
typedef __attribute__((ext_vector_type(4)))  float    v4f;
typedef __attribute__((ext_vector_type(8)))  float    v8f;

#define WMMA_F32_F16(a, b, c) \
    __builtin_amdgcn_wmma_f32_16x16x32_f16(false, (a), false, (b), (short)0, (c), false, false)

// ---------------------------------------------------------------------------
// WMMA fragment gathers from LDS (layouts per CDNA5 ISA 7.12.2)
// A (16x32 f16): lane m (m<16) holds row m, K {0..7,16..23}; lane m+16 holds
//                row m, K {8..15, 24..31}. Two ds_load_b128 per lane.
// B (32x16 f16): lane n (n<16) holds col n, K {0..7,16..23}; lane n+16 col n, K {8..15,24..31}.
// C/D (16x16 f32): vgpr i: lanes 0-15 -> (M=i, N=lane); lanes 16-31 -> (M=i+8).
// ---------------------------------------------------------------------------
__device__ inline v16h frag_a(const h16* base, int ld, int mBase, int lane) {
    const int lm = lane & 15, lh = lane >> 4;
    const h16* p = base + (size_t)(mBase + lm) * ld;
    v8h lo = *(const v8h*)(p + lh * 8);
    v8h hi = *(const v8h*)(p + 16 + lh * 8);
    v16h r;
#pragma unroll
    for (int i = 0; i < 8; ++i) { r[i] = lo[i]; r[8 + i] = hi[i]; }
    return r;
}

__device__ inline v16h frag_b(const h16* base, int ld, int nBase, int lane) {
    const int lm = lane & 15, lh = lane >> 4;
    const h16* p = base + (size_t)(nBase + lm) * ld;
    v8h lo = *(const v8h*)(p + lh * 8);
    v8h hi = *(const v8h*)(p + 16 + lh * 8);
    v16h r;
#pragma unroll
    for (int i = 0; i < 8; ++i) { r[i] = lo[i]; r[8 + i] = hi[i]; }
    return r;
}

// ---------------------------------------------------------------------------
// Generic GEMM: C[M,N] = act(A[M,K] * W[K,N] + bias[N])
// MODE 0: bias only.  MODE 1: elu(x)+1 feature map after bias.
// 128x128 block, BK=32, 8 waves (2x4), each wave 64x32 = 4x2 WMMA tiles.
// ---------------------------------------------------------------------------
template <typename TA, typename TO, int MODE>
__global__ __launch_bounds__(256) void gemm_act(const TA* __restrict__ A,
                                                const float* __restrict__ W,
                                                const float* __restrict__ bias,
                                                TO* __restrict__ C,
                                                int M, int K, int N) {
    __shared__ __align__(16) h16 As[128 * 40];
    __shared__ __align__(16) h16 Bs[128 * 40];
    const int tid  = threadIdx.x;
    const int row0 = blockIdx.x * 128;
    const int n0   = blockIdx.y * 128;
    const int wave = tid >> 5, lane = tid & 31;
    const int wm = (wave >> 2) * 64;   // 0 or 64
    const int wn = (wave & 3) * 32;    // 0,32,64,96

    // staging maps
    const int ar   = tid >> 1;              // A: row 0..127
    const int akof = (tid & 1) * 16;        // A: k-offset 0 or 16
    const int bkk  = (tid & 7) * 4;         // B: 4 consecutive k
    const int bng  = (tid >> 3) * 4;        // B: 4 consecutive n

    const v8f vzero = {0.f, 0.f, 0.f, 0.f, 0.f, 0.f, 0.f, 0.f};
    v8f acc[4][2];
#pragma unroll
    for (int i = 0; i < 4; ++i)
#pragma unroll
        for (int j = 0; j < 2; ++j) acc[i][j] = vzero;

    for (int k0 = 0; k0 < K; k0 += 32) {
        // stage A tile (128x32) row-major, explicit b128 LDS stores
        {
            const TA* src = A + (size_t)(row0 + ar) * K + k0 + akof;
            h16* dst = As + ar * 40 + akof;
            v8h o0, o1;
#pragma unroll
            for (int i = 0; i < 8; ++i) {
                o0[i] = (h16)(float)src[i];
                o1[i] = (h16)(float)src[8 + i];
            }
            *(v8h*)(dst) = o0;
            *(v8h*)(dst + 8) = o1;
        }
        // stage B tile (32x128) column-major; pack k-quads -> ds_store_b64
        {
            const float* src = W + (size_t)(k0 + bkk) * N + n0 + bng;
            v4f w0 = *(const v4f*)(src);
            v4f w1 = *(const v4f*)(src + N);
            v4f w2 = *(const v4f*)(src + 2 * N);
            v4f w3 = *(const v4f*)(src + 3 * N);
#pragma unroll
            for (int n = 0; n < 4; ++n) {
                v4h pk = {(h16)w0[n], (h16)w1[n], (h16)w2[n], (h16)w3[n]};
                *(v4h*)(Bs + (bng + n) * 40 + bkk) = pk;
            }
        }
        // prefetch next tiles into cache while this one is consumed
        if (k0 + 32 < K) {
            __builtin_prefetch(A + (size_t)(row0 + ar) * K + k0 + 32 + akof, 0, 3);
            __builtin_prefetch(W + (size_t)(k0 + 32 + bkk) * N + n0 + bng, 0, 3);
        }
        __syncthreads();

        v16h af[4], bf[2];
#pragma unroll
        for (int i = 0; i < 4; ++i) af[i] = frag_a(As, 40, wm + i * 16, lane);
#pragma unroll
        for (int j = 0; j < 2; ++j) bf[j] = frag_b(Bs, 40, wn + j * 16, lane);
#pragma unroll
        for (int i = 0; i < 4; ++i)
#pragma unroll
            for (int j = 0; j < 2; ++j)
                acc[i][j] = WMMA_F32_F16(af[i], bf[j], acc[i][j]);
        __syncthreads();
    }

    const int lm = lane & 15, lh = lane >> 4;
#pragma unroll
    for (int i = 0; i < 4; ++i)
#pragma unroll
        for (int j = 0; j < 2; ++j)
#pragma unroll
            for (int e = 0; e < 8; ++e) {
                const int rg = row0 + wm + i * 16 + lh * 8 + e;
                const int cg = n0 + wn + j * 16 + lm;
                float val = acc[i][j][e] + bias[cg];
                if (MODE == 1) val = (val > 0.f) ? (val + 1.f) : __expf(val);
                C[(size_t)rg * N + cg] = (TO)val;
            }
}

// ---------------------------------------------------------------------------
// kv partials: for (b,h,kchunk): kvp[d,m] = sum_{s in chunk} kf[s,d]*v[s,m]
// M=64 (d), N=64 (m), split-K 16 x 512. Deterministic (no atomics).
// Staging packs s-pairs so transposed LDS writes are b32, not b16.
// ---------------------------------------------------------------------------
__global__ __launch_bounds__(256) void kv_kernel(const h16* __restrict__ kf,
                                                 const h16* __restrict__ vv,
                                                 float* __restrict__ kvp,
                                                 int L) {
    __shared__ __align__(16) h16 As[64 * 40];  // kf transposed: [d][s]
    __shared__ __align__(16) h16 Bs[64 * 40];  // v col-major:   [m][s]
    const int tid = threadIdx.x;
    const int bh = blockIdx.x >> 4;   // 0..63
    const int kc = blockIdx.x & 15;   // split-K chunk
    const int b = bh >> 4, h = bh & 15;
    const int sBase0 = kc * 512;
    const int wave = tid >> 5, lane = tid & 31;
    const int dT = (wave >> 1) * 16;   // 0,16,32,48
    const int mB = (wave & 1) * 32;    // 0 or 32

    const int sl = (tid & 15) * 2;     // s pair
    const int dg = (tid >> 4) * 4;     // 4 d's

    const v8f vzero = {0.f, 0.f, 0.f, 0.f, 0.f, 0.f, 0.f, 0.f};
    v8f acc[2] = {vzero, vzero};

    for (int ks = 0; ks < 512; ks += 32) {
        const size_t g0 = ((size_t)b * L + sBase0 + ks + sl) * 1024 + h * 64 + dg;
        v4h ka = *(const v4h*)(kf + g0);
        v4h kb = *(const v4h*)(kf + g0 + 1024);
        v4h va = *(const v4h*)(vv + g0);
        v4h vb = *(const v4h*)(vv + g0 + 1024);
#pragma unroll
        for (int i = 0; i < 4; ++i) {
            v2h pk = {ka[i], kb[i]};
            *(v2h*)(As + (dg + i) * 40 + sl) = pk;
            v2h pv = {va[i], vb[i]};
            *(v2h*)(Bs + (dg + i) * 40 + sl) = pv;
        }
        __syncthreads();
        v16h a  = frag_a(As, 40, dT, lane);
        v16h b0 = frag_b(Bs, 40, mB, lane);
        v16h b1 = frag_b(Bs, 40, mB + 16, lane);
        acc[0] = WMMA_F32_F16(a, b0, acc[0]);
        acc[1] = WMMA_F32_F16(a, b1, acc[1]);
        __syncthreads();
    }

    const int lm = lane & 15, lh = lane >> 4;
    float* out = kvp + ((size_t)kc * 64 + bh) * 4096;
#pragma unroll
    for (int j = 0; j < 2; ++j)
#pragma unroll
        for (int e = 0; e < 8; ++e) {
            const int d = dT + lh * 8 + e;
            const int m = mB + j * 16 + lm;
            out[d * 64 + m] = acc[j][e];
        }
}

// ksum partials: ksum_part[p][b*1024+c] = sum of 512 rows of kf column c
__global__ __launch_bounds__(256) void ksum_kernel(const h16* __restrict__ kf,
                                                   float* __restrict__ ksum_part,
                                                   int L) {
    const int b = blockIdx.x >> 4;
    const int rc = blockIdx.x & 15;
    const int c = blockIdx.y * 256 + threadIdx.x;
    float s = 0.f;
    const size_t base = ((size_t)b * L + (size_t)rc * 512) * 1024 + c;
    for (int i = 0; i < 512; ++i) s += (float)kf[base + (size_t)i * 1024];
    ksum_part[(size_t)rc * 4096 + b * 1024 + c] = s;
}

// out[i] = sum_{p<16} in[p*n + i]  (fixed order -> deterministic)
__global__ void reduce16(const float* __restrict__ in, float* __restrict__ out, int n) {
    const int g = blockIdx.x * 256 + threadIdx.x;
    if (g >= n) return;
    float s = 0.f;
#pragma unroll
    for (int p = 0; p < 16; ++p) s += in[(size_t)p * n + g];
    out[g] = s;
}

// z[n*16+h] = 1 / (dot(qf[n,h,:], ksum[b,h,:]) + eps)
__global__ void z_kernel(const h16* __restrict__ qf, const float* __restrict__ ksum,
                         float* __restrict__ z, int NH) {
    const int g = blockIdx.x * 256 + threadIdx.x;
    if (g >= NH) return;
    const int n = g >> 4, h = g & 15;
    const int b = n >> 13;  // L = 8192
    const h16* qp = qf + (size_t)n * 1024 + h * 64;
    const float* kp = ksum + b * 1024 + h * 64;
    float s = 0.f;
#pragma unroll
    for (int d = 0; d < 64; ++d) s += (float)qp[d] * kp[d];
    z[g] = 1.0f / (s + 1e-6f);
}

// y = (qf @ kv^T) * z, per (b,h): [256x64] x [64x64] per WG, z fused at store.
// qf staging uses gfx1250 async memory->LDS copies (ASYNCcnt tracked).
__global__ __launch_bounds__(256) void y_kernel(const h16* __restrict__ qf,
                                                const float* __restrict__ kvT,
                                                const float* __restrict__ z,
                                                h16* __restrict__ y, int L) {
    __shared__ __align__(16) h16 As[256 * 72];
    __shared__ __align__(16) h16 Bs[64 * 72];
    const int tid = threadIdx.x;
    const int bh = blockIdx.x;
    const int b = bh >> 4, h = bh & 15;
    const int row0 = b * L + blockIdx.y * 256;

    {   // stage qf rows: 128 contiguous bytes per row, async global->LDS
        const h16* src = qf + (size_t)(row0 + tid) * 1024 + h * 64;
        const unsigned ldsoff = (unsigned)(uintptr_t)(As + tid * 72);  // LDS aperture: addr[31:0]
        const unsigned long long ga = (unsigned long long)(uintptr_t)src;
        asm volatile(
            "global_load_async_to_lds_b128 %0, %1, off\n\t"
            "global_load_async_to_lds_b128 %0, %1, off offset:16\n\t"
            "global_load_async_to_lds_b128 %0, %1, off offset:32\n\t"
            "global_load_async_to_lds_b128 %0, %1, off offset:48\n\t"
            "global_load_async_to_lds_b128 %0, %1, off offset:64\n\t"
            "global_load_async_to_lds_b128 %0, %1, off offset:80\n\t"
            "global_load_async_to_lds_b128 %0, %1, off offset:96\n\t"
            "global_load_async_to_lds_b128 %0, %1, off offset:112\n\t"
            :
            : "v"(ldsoff), "v"(ga)
            : "memory");
    }
    {   // stage kv^T (f32 [d][m]) -> LDS col-major f16 Bs[m][d]
        const float* src = kvT + (size_t)bh * 4096 + tid * 16;
#pragma unroll
        for (int i = 0; i < 16; ++i) {
            const int idx = tid * 16 + i;          // idx = d*64 + m
            Bs[(idx & 63) * 72 + (idx >> 6)] = (h16)src[i];
        }
    }
    asm volatile("s_wait_asynccnt 0x0" ::: "memory");
    __syncthreads();

    const int wave = tid >> 5, lane = tid & 31;
    const int wm = wave * 32;
    const v8f vzero = {0.f, 0.f, 0.f, 0.f, 0.f, 0.f, 0.f, 0.f};
    v8f acc[2][4];
#pragma unroll
    for (int i = 0; i < 2; ++i)
#pragma unroll
        for (int j = 0; j < 4; ++j) acc[i][j] = vzero;

#pragma unroll
    for (int ks = 0; ks < 2; ++ks) {
        v16h a0 = frag_a(As + ks * 32, 72, wm, lane);
        v16h a1 = frag_a(As + ks * 32, 72, wm + 16, lane);
#pragma unroll
        for (int j = 0; j < 4; ++j) {
            v16h bb = frag_b(Bs + ks * 32, 72, j * 16, lane);
            acc[0][j] = WMMA_F32_F16(a0, bb, acc[0][j]);
            acc[1][j] = WMMA_F32_F16(a1, bb, acc[1][j]);
        }
    }

    const int lm = lane & 15, lh = lane >> 4;
#pragma unroll
    for (int i = 0; i < 2; ++i)
#pragma unroll
        for (int j = 0; j < 4; ++j)
#pragma unroll
            for (int e = 0; e < 8; ++e) {
                const int rg = row0 + wm + i * 16 + lh * 8 + e;
                const int m = j * 16 + lm;
                const float zv = z[rg * 16 + h];
                y[(size_t)rg * 1024 + h * 64 + m] = (h16)(acc[i][j][e] * zv);
            }
}

// ---------------------------------------------------------------------------
extern "C" void kernel_launch(void* const* d_in, const int* in_sizes, int n_in,
                              void* d_out, int out_size, void* d_ws, size_t ws_size,
                              hipStream_t stream) {
    (void)in_sizes; (void)n_in; (void)out_size; (void)ws_size;
    const float* x  = (const float*)d_in[0];
    const float* Wq = (const float*)d_in[1];
    const float* bq = (const float*)d_in[2];
    const float* Wk = (const float*)d_in[3];
    const float* bk = (const float*)d_in[4];
    const float* Wv = (const float*)d_in[5];
    const float* bv = (const float*)d_in[6];
    const float* Wo = (const float*)d_in[7];
    const float* bo = (const float*)d_in[8];

    const int Bb = 4, L = 8192, D = 1024, H = 16;
    const int N = Bb * L;                    // 32768
    const size_t NDh = (size_t)N * D * sizeof(h16);   // 64 MiB

    char* ws = (char*)d_ws;
    size_t off = 0;
    h16* qf = (h16*)(ws + off); off += NDh;
    h16* kf = (h16*)(ws + off); off += NDh;
    h16* vv = (h16*)(ws + off); off += NDh;
    h16* yy = (h16*)(ws + off); off += NDh;
    float* kvp   = (float*)(ws + off); off += (size_t)16 * 64 * 4096 * sizeof(float);
    float* kvT   = (float*)(ws + off); off += (size_t)64 * 4096 * sizeof(float);
    float* kspar = (float*)(ws + off); off += (size_t)16 * 4096 * sizeof(float);
    float* ksum  = (float*)(ws + off); off += (size_t)4096 * sizeof(float);
    float* zz    = (float*)(ws + off); off += (size_t)N * H * sizeof(float);

    const dim3 gProj(N / 128, D / 128);  // (256, 8)

    // QKV projections with fused bias (+ elu+1 feature map for q,k)
    gemm_act<float, h16, 1><<<gProj, 256, 0, stream>>>(x, Wq, bq, qf, N, D, D);
    gemm_act<float, h16, 1><<<gProj, 256, 0, stream>>>(x, Wk, bk, kf, N, D, D);
    gemm_act<float, h16, 0><<<gProj, 256, 0, stream>>>(x, Wv, bv, vv, N, D, D);

    // ksum = column sums of kf (split + deterministic reduce)
    ksum_kernel<<<dim3(Bb * 16, 4), 256, 0, stream>>>(kf, kspar, L);
    reduce16<<<(4096 + 255) / 256, 256, 0, stream>>>(kspar, ksum, 4096);

    // kv^T[d,m] = sum_s kf[s,d]*v[s,m] per (b,h) (split-K + deterministic reduce)
    kv_kernel<<<64 * 16, 256, 0, stream>>>(kf, vv, kvp, L);
    reduce16<<<(64 * 4096 + 255) / 256, 256, 0, stream>>>(kvp, kvT, 64 * 4096);

    // normalization
    z_kernel<<<(N * H + 255) / 256, 256, 0, stream>>>(qf, ksum, zz, N * H);

    // y = (qf @ kv^T) * z
    y_kernel<<<dim3(64, L / 256), 256, 0, stream>>>(qf, kvT, zz, yy, L);

    // output projection to f32 d_out
    gemm_act<h16, float, 0><<<gProj, 256, 0, stream>>>(yy, Wo, bo, (float*)d_out, N, D, D);
}
